// GraphConvLayer_88484916232487
// MI455X (gfx1250) — matrix operand order repaired and checked by reading, not compiled, not run on hardware
//
#include <hip/hip_runtime.h>
#include <hip/hip_bf16.h>
#include <math.h>
#include <stdint.h>

#define N_NODES 512
#define IN_DIM  128
#define OUT_DIM 128
#define EDGE_DIM 64
#define EDGE_THR 0.1f

typedef __attribute__((ext_vector_type(16))) _Float16 v16h;
typedef __attribute__((ext_vector_type(8)))  _Float16 v8h;
typedef __attribute__((ext_vector_type(8)))  float    v8f;
typedef int v4i_b128 __attribute__((vector_size(16)));

#define ASG __attribute__((address_space(1)))
#define ASL __attribute__((address_space(3)))

#if defined(__AMDGCN__) && \
    __has_builtin(__builtin_amdgcn_global_load_async_to_lds_b128) && \
    __has_builtin(__builtin_amdgcn_s_wait_asynccnt)
#define USE_ASYNC_LDS 1
#else
#define USE_ASYNC_LDS 0
#endif

// ---------------------------------------------------------------------------
// Fused directed-message kernel with double-buffered async LDS staging.
//   A_part[chunk][i,h] = sum_{j in chunk} wd[j,i] *
//                        relu( ef[j,i,:] @ we + src[j,h] + tgt[i,h] + db1[h] )
// 8 waves; wave w owns h-tile w and iterates the 16 targets.
// ---------------------------------------------------------------------------
__global__ __launch_bounds__(256) void edge_dir_kernel(
    const float* __restrict__ ef,   // [512][512][64]
    const float* __restrict__ W,    // [512][512]
    const float* __restrict__ we,   // dir_w1 + 2*IN_DIM*OUT_DIM : [64][128]
    const float* __restrict__ src,  // [512][128]
    const float* __restrict__ tgt,  // [512][128]
    const float* __restrict__ db1,  // [128]
    float* __restrict__ A_part)     // [8][512][128]
{
  __shared__ __align__(16) float lds_ef[2][16 * 1024];  // 2 x 64 KB [jj][il][e]

  const int i0   = blockIdx.x * 16;
  const int jc0  = blockIdx.y * 64;
  const int tid  = threadIdx.x;
  const int wave = tid >> 5;           // h-tile index 0..7
  const int lane = tid & 31;
  const int ln16 = lane & 15;
  const int hi   = lane >> 4;
  const int h0   = wave * 16;

  // Stage one 16-j x 16-i x 64-e f32 sub-tile into LDS buffer `buf`.
  auto stage = [&](int buf, int j0s) {
    float* dst = lds_ef[buf];
#if USE_ASYNC_LDS
#pragma unroll
    for (int jj = 0; jj < 16; ++jj) {
      const float* g = ef + (size_t)(j0s + jj) * (N_NODES * EDGE_DIM)
                          + (size_t)i0 * EDGE_DIM + tid * 4;
      __builtin_amdgcn_global_load_async_to_lds_b128(
          (ASG v4i_b128*)(uintptr_t)g,
          (ASL v4i_b128*)(uint32_t)(uintptr_t)(dst + jj * 1024 + tid * 4),
          0, 0);
    }
#else
#pragma unroll
    for (int jj = 0; jj < 16; ++jj) {
      const float4 v = ((const float4*)(ef + (size_t)(j0s + jj) * (N_NODES * EDGE_DIM)
                                           + (size_t)i0 * EDGE_DIM))[tid];
      *(float4*)(dst + jj * 1024 + tid * 4) = v;
    }
#endif
  };

  // Hoist B fragments (edge-weight slice of dir_w1) for this wave's h-tile.
  v16h bfrag[2];
#pragma unroll
  for (int ks = 0; ks < 2; ++ks) {
    const int kb = ks * 32 + hi * 16;
    v16h b;
#pragma unroll
    for (int e = 0; e < 16; ++e)
      b[e] = (_Float16)we[(size_t)(kb + e) * OUT_DIM + h0 + ln16];
    bfrag[ks] = b;
  }
  const float db1v = db1[h0 + ln16];

  float accD[16];
#pragma unroll
  for (int il = 0; il < 16; ++il) accD[il] = 0.0f;

  stage(0, jc0);
#if USE_ASYNC_LDS
  __builtin_amdgcn_s_wait_asynccnt(0);
#endif
  __syncthreads();

  for (int js = 0; js < 4; ++js) {
    const int j0 = jc0 + js * 16;
    const float* cur = lds_ef[js & 1];
    if (js < 3) stage((js + 1) & 1, j0 + 16);  // prefetch next sub-tile

    // src values for this half-wave's 8 j-rows at column h0+ln16
    float srcv[8];
#pragma unroll
    for (int r = 0; r < 8; ++r)
      srcv[r] = src[(size_t)(j0 + hi * 8 + r) * OUT_DIM + h0 + ln16];

    for (int il = 0; il < 16; ++il) {
      float wdv[8];
#pragma unroll
      for (int r = 0; r < 8; ++r) {
        float w = fabsf(W[(size_t)(j0 + hi * 8 + r) * N_NODES + (i0 + il)]);
        wdv[r] = (w > EDGE_THR) ? w : 0.0f;
      }
      const float tb = tgt[(size_t)(i0 + il) * OUT_DIM + h0 + ln16] + db1v;
      v8f c;
#pragma unroll
      for (int r = 0; r < 8; ++r) c[r] = srcv[r] + tb;

#pragma unroll
      for (int ks = 0; ks < 2; ++ks) {
        const float* ap = &cur[(ln16 * 16 + il) * 64 + ks * 32 + hi * 16];
        v16h a;
#pragma unroll
        for (int e = 0; e < 8; ++e) {
          a[e]     = (_Float16)ap[e];        // v_cvt_pk_f16_f32 pairs
          a[e + 8] = (_Float16)ap[e + 8];
        }
        c = __builtin_amdgcn_wmma_f32_16x16x32_f16(false, a, false, bfrag[ks],
                                                   (short)0, c, false, false);
      }
      float part = 0.0f;
#pragma unroll
      for (int r = 0; r < 8; ++r) part += fmaxf(c[r], 0.0f) * wdv[r];
      accD[il] += part;
    }
#if USE_ASYNC_LDS
    __builtin_amdgcn_s_wait_asynccnt(0);
#endif
    __syncthreads();
  }

  // Combine the two lane-halves (rows 0-7 vs 8-15) and store partials.
  float* out = A_part + (size_t)blockIdx.y * (N_NODES * OUT_DIM);
#pragma unroll
  for (int il = 0; il < 16; ++il) {
    float v = accD[il];
    v += __shfl_xor(v, 16, 32);
    if (lane < 16) out[(size_t)(i0 + il) * OUT_DIM + h0 + lane] = v;
  }
}

// ---------------------------------------------------------------------------
// Generic 16x16-tile WMMA GEMM (K fixed at 128): C = op(A[M,128]@B[128,128]+bias)
// One wave per 16x16 tile; 4 fully-unrolled v_wmma_f32_16x16x32_f16.
// ---------------------------------------------------------------------------
__global__ __launch_bounds__(32) void gemm_wmma_kernel(
    const float* __restrict__ A, const float* __restrict__ B,
    const float* __restrict__ bias, float* __restrict__ C, int do_relu)
{
  const int m0   = blockIdx.x * 16;
  const int n0   = blockIdx.y * 16;
  const int lane = threadIdx.x;
  const int ln16 = lane & 15;
  const int hi   = lane >> 4;

  v8f c;
  const float bv = bias ? bias[n0 + ln16] : 0.0f;
#pragma unroll
  for (int r = 0; r < 8; ++r) c[r] = bv;

#pragma unroll
  for (int kk = 0; kk < 128; kk += 32) {
    const float* ap = A + (size_t)(m0 + ln16) * 128 + kk + hi * 16;
    v16h a;
#pragma unroll
    for (int e = 0; e < 8; ++e) {
      a[e]     = (_Float16)ap[e];
      a[e + 8] = (_Float16)ap[e + 8];
    }
    const int kb = kk + hi * 16;
    v16h b;
#pragma unroll
    for (int e = 0; e < 16; ++e)
      b[e] = (_Float16)B[(size_t)(kb + e) * OUT_DIM + n0 + ln16];

    c = __builtin_amdgcn_wmma_f32_16x16x32_f16(false, a, false, b,
                                               (short)0, c, false, false);
  }

#pragma unroll
  for (int r = 0; r < 8; ++r) {
    float v = c[r];
    if (do_relu) v = fmaxf(v, 0.0f);
    C[(size_t)(m0 + hi * 8 + r) * OUT_DIM + n0 + ln16] = v;
  }
}

__global__ __launch_bounds__(256) void reduce_part_kernel(
    const float* __restrict__ part, float* __restrict__ A_dir)
{
  const int idx = blockIdx.x * 256 + threadIdx.x;  // 0..65535
  float s = 0.0f;
#pragma unroll
  for (int c = 0; c < 8; ++c) s += part[(size_t)c * (N_NODES * OUT_DIM) + idx];
  A_dir[idx] = s;
}

// ---------------------------------------------------------------------------
// Bidirected path (pure elementwise after factoring):
//   A_bi[i,h] = sum_{j != i} wb[i,j] * relu(bsrc[j,h] + btgt[i,h] + bb1[h])
// ---------------------------------------------------------------------------
__global__ __launch_bounds__(128) void bi_kernel(
    const float* __restrict__ Mm, const float* __restrict__ bsrc,
    const float* __restrict__ btgt, const float* __restrict__ bb1,
    float* __restrict__ A_bi)
{
  const int i0 = blockIdx.x * 8;
  const int t  = threadIdx.x;  // h
  float tb[8], acc[8];
#pragma unroll
  for (int u = 0; u < 8; ++u) {
    tb[u]  = btgt[(size_t)(i0 + u) * OUT_DIM + t] + bb1[t];
    acc[u] = 0.0f;
  }
  for (int j = 0; j < N_NODES; ++j) {
    const float s = bsrc[(size_t)j * OUT_DIM + t];
#pragma unroll
    for (int u = 0; u < 8; ++u) {
      const int i = i0 + u;
      float w = fabsf(Mm[(size_t)i * N_NODES + j]);
      w = (w > EDGE_THR && j != i) ? w : 0.0f;
      acc[u] += w * fmaxf(s + tb[u], 0.0f);
    }
  }
#pragma unroll
  for (int u = 0; u < 8; ++u) A_bi[(size_t)(i0 + u) * OUT_DIM + t] = acc[u];
}

// swd[i] = sum_j masked|W[j,i]| ; swb[i] = sum_{j!=i} masked|M[i,j]|
__global__ __launch_bounds__(256) void sums_kernel(
    const float* __restrict__ W, const float* __restrict__ Mm,
    float* __restrict__ swd, float* __restrict__ swb)
{
  const int i = blockIdx.x * 256 + threadIdx.x;
  if (i >= N_NODES) return;
  float sd = 0.0f, sb = 0.0f;
  for (int j = 0; j < N_NODES; ++j) {
    float w = fabsf(W[(size_t)j * N_NODES + i]);
    if (w > EDGE_THR) sd += w;
    float m = fabsf(Mm[(size_t)i * N_NODES + j]);
    if (m > EDGE_THR && j != i) sb += m;
  }
  swd[i] = sd;
  swb[i] = sb;
}

// h = h_self + (A_dir@w2 + swd*b2) + (A_bi@w2 + swb*b2); LayerNorm(h)
__global__ __launch_bounds__(128) void ln_kernel(
    const float* __restrict__ h_self, const float* __restrict__ tmp_dir,
    const float* __restrict__ tmp_bi, const float* __restrict__ swd,
    const float* __restrict__ swb, const float* __restrict__ db2,
    const float* __restrict__ bb2, const float* __restrict__ ln_g,
    const float* __restrict__ ln_b, float* __restrict__ out)
{
  const int i = blockIdx.x;
  const int t = threadIdx.x;
  const size_t idx = (size_t)i * OUT_DIM + t;
  float v = h_self[idx] + tmp_dir[idx] + swd[i] * db2[t]
          + tmp_bi[idx] + swb[i] * bb2[t];

  __shared__ float rs[4], rq[4];
  float s = v;
#pragma unroll
  for (int o = 16; o > 0; o >>= 1) s += __shfl_xor(s, o, 32);
  if ((t & 31) == 0) rs[t >> 5] = s;
  __syncthreads();
  const float mean = (rs[0] + rs[1] + rs[2] + rs[3]) * (1.0f / OUT_DIM);
  const float d = v - mean;
  float q = d * d;
#pragma unroll
  for (int o = 16; o > 0; o >>= 1) q += __shfl_xor(q, o, 32);
  if ((t & 31) == 0) rq[t >> 5] = q;
  __syncthreads();
  const float var = (rq[0] + rq[1] + rq[2] + rq[3]) * (1.0f / OUT_DIM);
  out[idx] = d * rsqrtf(var + 1e-5f) * ln_g[t] + ln_b[t];
}

// ---------------------------------------------------------------------------
extern "C" void kernel_launch(void* const* d_in, const int* in_sizes, int n_in,
                              void* d_out, int out_size, void* d_ws, size_t ws_size,
                              hipStream_t stream) {
  (void)in_sizes; (void)n_in; (void)out_size; (void)ws_size;
  const float* x       = (const float*)d_in[0];
  const float* W       = (const float*)d_in[1];
  const float* Mm      = (const float*)d_in[2];
  const float* ef      = (const float*)d_in[3];
  const float* node_w1 = (const float*)d_in[4];
  const float* node_b1 = (const float*)d_in[5];
  const float* node_w2 = (const float*)d_in[6];
  const float* node_b2 = (const float*)d_in[7];
  const float* dir_w1  = (const float*)d_in[8];
  const float* dir_b1  = (const float*)d_in[9];
  const float* dir_w2  = (const float*)d_in[10];
  const float* dir_b2  = (const float*)d_in[11];
  const float* bi_w1   = (const float*)d_in[12];
  const float* bi_b1   = (const float*)d_in[13];
  const float* bi_w2   = (const float*)d_in[14];
  const float* bi_b2   = (const float*)d_in[15];
  const float* ln_g    = (const float*)d_in[16];
  const float* ln_b    = (const float*)d_in[17];

  float* ws = (float*)d_ws;
  const size_t NM = (size_t)N_NODES * OUT_DIM;  // 65536
  float* srcv   = ws;
  float* tgtv   = ws + 1 * NM;
  float* bsrc   = ws + 2 * NM;
  float* btgt   = ws + 3 * NM;
  float* t1     = ws + 4 * NM;
  float* h_self = ws + 5 * NM;
  float* A_dir  = ws + 6 * NM;
  float* A_bi   = ws + 7 * NM;
  float* tmpd   = ws + 8 * NM;
  float* tmpb   = ws + 9 * NM;
  float* swd    = ws + 10 * NM;
  float* swb    = swd + N_NODES;
  float* A_part = swb + N_NODES;  // [8][512][128]

  const dim3 gg(N_NODES / 16, OUT_DIM / 16);  // (32, 8), 1 wave per tile

  gemm_wmma_kernel<<<gg, 32, 0, stream>>>(x, dir_w1,                  nullptr, srcv, 0);
  gemm_wmma_kernel<<<gg, 32, 0, stream>>>(x, dir_w1 + IN_DIM*OUT_DIM, nullptr, tgtv, 0);
  gemm_wmma_kernel<<<gg, 32, 0, stream>>>(x, bi_w1,                   nullptr, bsrc, 0);
  gemm_wmma_kernel<<<gg, 32, 0, stream>>>(x, bi_w1 + IN_DIM*OUT_DIM,  nullptr, btgt, 0);
  gemm_wmma_kernel<<<gg, 32, 0, stream>>>(x, node_w1, node_b1, t1, 1);
  gemm_wmma_kernel<<<gg, 32, 0, stream>>>(t1, node_w2, node_b2, h_self, 0);

  sums_kernel<<<2, 256, 0, stream>>>(W, Mm, swd, swb);

  edge_dir_kernel<<<dim3(N_NODES / 16, 8), 256, 0, stream>>>(
      ef, W, dir_w1 + 2 * IN_DIM * OUT_DIM, srcv, tgtv, dir_b1, A_part);
  reduce_part_kernel<<<256, 256, 0, stream>>>(A_part, A_dir);

  bi_kernel<<<N_NODES / 8, 128, 0, stream>>>(Mm, bsrc, btgt, bi_b1, A_bi);

  gemm_wmma_kernel<<<gg, 32, 0, stream>>>(A_dir, dir_w2, nullptr, tmpd, 0);
  gemm_wmma_kernel<<<gg, 32, 0, stream>>>(A_bi,  bi_w2,  nullptr, tmpb, 0);

  ln_kernel<<<N_NODES, 128, 0, stream>>>(h_self, tmpd, tmpb, swd, swb,
                                         dir_b2, bi_b2, ln_g, ln_b, (float*)d_out);
}
